// ChessDiT_55972013802289
// MI455X (gfx1250) — matrix operand order repaired and checked by reading, not compiled, not run on hardware
//
#include <hip/hip_runtime.h>
#include <hip/hip_bf16.h>
#include <math.h>
#include <stdint.h>

typedef __bf16 bf16;
typedef __attribute__((ext_vector_type(8)))  bf16  v8bf;
typedef __attribute__((ext_vector_type(16))) bf16  v16bf;
typedef __attribute__((ext_vector_type(8)))  float v8f;

#define B_  4096
#define S_  64
#define D_  256
#define DM_ 512
#define H_  4
#define DH_ 64
#define DF_ 512
#define L_  6
#define MODN 9728   // 6*1536 adaLN + 512 final = combined modulation width

// ---- dynamic LDS layout (bytes), all offsets 16B aligned ----
#define HS 264     // row stride (elems) for hb/qb/kb  (256 + 8 pad)
#define VS 72      // row stride for vt (v transposed [d][s])
#define PS 72      // row stride for probs
#define FS 520     // row stride for ffn hidden (512 + 8 pad)

#define XF_OFF   0        // float x[64][256]            65536
#define HB_OFF   65536    // bf16 hb[64][HS]             33792
#define QB_OFF   99328    // bf16 q [64][HS]             33792
#define KB_OFF   133120   // bf16 k [64][HS]             33792
#define VT_OFF   166912   // bf16 vt[256][VS]            36864
#define PR_OFF   203776   // bf16 probs[4][64][PS]       36864
#define MOD_OFF  240640   // float modv[1536]             6144
#define FFB_OFF  QB_OFF   // bf16 ffb[64][FS] aliases q/k/vt (66560 <= 104448)
#define SMEM_BYTES 246784

__device__ __forceinline__ float mish_f(float x) {
  float sp = (x > 20.f) ? x : log1pf(__expf(x));
  return x * tanhf(sp);
}

__device__ __forceinline__ v16bf frag_cat(v8bf lo, v8bf hi) {
  v16bf r;
#pragma unroll
  for (int i = 0; i < 8; ++i) { r[i] = lo[i]; r[8 + i] = hi[i]; }
  return r;
}

// A fragment: 16x32 bf16, A row-major [M][ks]. Per ISA layout:
// lanes 0-15 (half=0): VGPR0-3 = K0..7,  VGPR4-7 = K16..23
// lanes16-31 (half=1): VGPR0-3 = K8..15, VGPR4-7 = K24..31
__device__ __forceinline__ v16bf load_a(const bf16* A, int mbase, int ks, int kk, int lane) {
  int half = (lane >> 4) & 1, m = lane & 15;
  const bf16* p = A + (size_t)(mbase + m) * ks + kk;
  v8bf lo = *(const v8bf*)(p + 8 * half);
  v8bf hi = *(const v8bf*)(p + 16 + 8 * half);
  return frag_cat(lo, hi);
}

// B fragment: 32x16 bf16 from Bt stored [N][ks] (B transposed).
// lanes 0-15: K0..7, K16..23; lanes 16-31: K8..15, K24..31 (same as A).
__device__ __forceinline__ v16bf load_b(const bf16* Bt, int nbase, int ks, int kk, int lane) {
  int half = (lane >> 4) & 1, n = lane & 15;
  const bf16* p = Bt + (size_t)(nbase + n) * ks + kk;
  v8bf lo = *(const v8bf*)(p + 8 * half);
  v8bf hi = *(const v8bf*)(p + 16 + 8 * half);
  return frag_cat(lo, hi);
}

__device__ __forceinline__ v8f wmma_bf16(v16bf a, v16bf b, v8f c) {
  return __builtin_amdgcn_wmma_f32_16x16x32_bf16(false, a, false, b, (short)0, c, false, false);
}

__device__ __forceinline__ v8f gemm_tile(const bf16* A, int mbase, int aks, int akofs,
                                         const bf16* Bt, int nbase, int bks, int bkofs,
                                         int K, int lane) {
  v8f acc = {};
  for (int kk = 0; kk < K; kk += 32) {
    v16bf a = load_a(A, mbase, aks, akofs + kk, lane);
    v16bf b = load_b(Bt, nbase, bks, bkofs + kk, lane);
    acc = wmma_bf16(a, b, acc);
  }
  return acc;
}

// async global->LDS copy (gfx1250 ASYNCcnt path)
__device__ __forceinline__ void async_g2l_b128(uint32_t lds_off, const void* g) {
  asm volatile("global_load_async_to_lds_b128 %0, %1, off" :: "v"(lds_off), "v"(g) : "memory");
}

// LayerNorm + adaLN modulate: dst = LN(x)*(1+sc)+sh, bf16 out
__device__ __forceinline__ void ln_mod(const float* xf, bf16* dst, int ds,
                                       const float* sc, const float* sh,
                                       int wave, int lane) {
  for (int s = wave; s < S_; s += 8) {
    const float* xr = xf + s * D_;
    float sum = 0.f, sq = 0.f;
    for (int k = lane; k < D_; k += 32) { float v = xr[k]; sum += v; sq += v * v; }
#pragma unroll
    for (int off = 16; off; off >>= 1) {
      sum += __shfl_xor(sum, off, 32);
      sq  += __shfl_xor(sq,  off, 32);
    }
    float mu = sum * (1.f / D_);
    float rstd = rsqrtf(sq * (1.f / D_) - mu * mu + 1e-5f);
    for (int k = lane; k < D_; k += 32) {
      float v = (xr[k] - mu) * rstd;
      dst[s * ds + k] = (bf16)(v * (1.f + sc[k]) + sh[k]);
    }
  }
}

// ---------------- weight prep: fp32 [mat][K][N] -> bf16 [mat][N][K] ----------------
__global__ void wt_prep(const float* __restrict__ src, bf16* __restrict__ dst,
                        int rows, int cols, int total) {
  int idx = blockIdx.x * blockDim.x + threadIdx.x;
  if (idx >= total) return;
  int per = rows * cols;
  int mat = idx / per, rem = idx - mat * per;
  int k = rem / cols, n = rem - k * cols;
  dst[(size_t)mat * per + (size_t)n * rows + k] = (bf16)src[idx];
}

// combined modulation weight: Wt[MODN][256] bf16 + bias[MODN] f32
__global__ void modw_prep(const float* __restrict__ adaln_w, const float* __restrict__ adaln_b,
                          const float* __restrict__ fa_w, const float* __restrict__ fa_b,
                          bf16* __restrict__ Wt, float* __restrict__ biasv) {
  int idx = blockIdx.x * blockDim.x + threadIdx.x;
  if (idx >= MODN * D_) return;
  int n = idx >> 8, k = idx & 255;
  float w, bv;
  if (n < 9216) {
    int l = n / 1536, j = n - l * 1536;
    w  = adaln_w[(size_t)l * D_ * 1536 + (size_t)k * 1536 + j];
    bv = adaln_b[l * 1536 + j];
  } else {
    int j = n - 9216;
    w  = fa_w[k * (2 * D_) + j];
    bv = fa_b[j];
  }
  Wt[(size_t)n * D_ + k] = (bf16)w;
  if (k == 0) biasv[n] = bv;
}

// teg[b] = emb[t[b]] (bf16); cm[b] = mean_s latent[b,s,:] (bf16)
__global__ void gather_kernel(const int* __restrict__ t, const float* __restrict__ emb,
                              const float* __restrict__ backbone,
                              bf16* __restrict__ teg, bf16* __restrict__ cm) {
  int b = blockIdx.x, tid = threadIdx.x;
  teg[(size_t)b * D_ + tid] = (bf16)emb[(size_t)t[b] * D_ + tid];
  const float* bb = backbone + (size_t)b * S_ * DM_;
  for (int j = 0; j < 2; ++j) {
    int col = j * D_ + tid;
    float s = 0.f;
    for (int si = 0; si < S_; ++si) s += bb[si * DM_ + col];
    cm[(size_t)b * DM_ + col] = (bf16)(s * (1.f / S_));
  }
}

// ---------------- generic batched WMMA GEMM: out = act(A @ Bt^T + bias [+ addsrc]) ----------------
// A: bf16 [M][K] row-major; Bt: bf16 [N][K]; block tile 64x128, 8 waves x 4 tiles
__global__ void __launch_bounds__(256)
gemm_bf16(const bf16* __restrict__ A, const bf16* __restrict__ Bt,
          const float* __restrict__ bias, const float* __restrict__ addsrc,
          bf16* __restrict__ outb, float* __restrict__ outf,
          int K, int N, int act) {
  int bn = blockIdx.x * 128, bm = blockIdx.y * 64;
  int tid = threadIdx.x, lane = tid & 31, wave = tid >> 5;
  int lane15 = lane & 15, half = (lane >> 4) & 1;
  for (int t = wave; t < 32; t += 8) {
    int mt = t >> 3, nt = t & 7;
    v8f acc = gemm_tile(A, bm + mt * 16, K, 0, Bt, bn + nt * 16, K, 0, K, lane);
    int n = bn + nt * 16 + lane15;
    float bv = bias ? bias[n] : 0.f;
#pragma unroll
    for (int r = 0; r < 8; ++r) {
      int m = bm + mt * 16 + r + 8 * half;
      float v = acc[r] + bv;
      if (addsrc) v += addsrc[(size_t)m * N + n];
      if (act) v = mish_f(v);
      if (outb) outb[(size_t)m * N + n] = (bf16)v;
      if (outf) outf[(size_t)m * N + n] = v;
    }
  }
}

// ---------------- fused DiT: one block = one batch element ----------------
__global__ void __launch_bounds__(256, 1)
dit_main(const float* __restrict__ x_t, const bf16* __restrict__ modb,
         const bf16* __restrict__ qkv_wt, const float* __restrict__ qkv_b,
         const bf16* __restrict__ op_wt, const float* __restrict__ op_b,
         const bf16* __restrict__ f1_wt, const float* __restrict__ f1_b,
         const bf16* __restrict__ f2_wt, const float* __restrict__ f2_b,
         const bf16* __restrict__ fp_wt, const float* __restrict__ fp_b,
         float* __restrict__ out) {
  extern __shared__ char smem[];
  float* xf   = (float*)(smem + XF_OFF);
  bf16*  hb   = (bf16*)(smem + HB_OFF);
  bf16*  qb   = (bf16*)(smem + QB_OFF);
  bf16*  kb   = (bf16*)(smem + KB_OFF);
  bf16*  vt   = (bf16*)(smem + VT_OFF);
  bf16*  pr   = (bf16*)(smem + PR_OFF);
  bf16*  ffb  = (bf16*)(smem + FFB_OFF);
  float* modv = (float*)(smem + MOD_OFF);

  const int b = blockIdx.x, tid = threadIdx.x;
  const int lane = tid & 31, wave = tid >> 5;
  const int lane15 = lane & 15, half = (lane >> 4) & 1;

  // async-load residual x into LDS (ASYNCcnt path)
  {
    const char* src = (const char*)(x_t + (size_t)b * S_ * D_);
    uint32_t l0 = (uint32_t)(uintptr_t)(void*)(smem + XF_OFF);
    for (int i = tid; i < (S_ * D_ * 4) / 16; i += 256)
      async_g2l_b128(l0 + i * 16, src + (size_t)i * 16);
    asm volatile("s_wait_asynccnt 0" ::: "memory");
  }
  __syncthreads();

  for (int l = 0; l < L_; ++l) {
    const bf16* Wq = qkv_wt + (size_t)l * 3 * D_ * D_;
    const bf16* Wo = op_wt  + (size_t)l * D_ * D_;
    const bf16* W1 = f1_wt  + (size_t)l * DF_ * D_;
    const bf16* W2 = f2_wt  + (size_t)l * D_ * DF_;
    __builtin_prefetch(Wq, 0, 1);
    __builtin_prefetch(W1, 0, 1);

    // (1) load precomputed adaLN modulation row
    {
      const bf16* mrow = modb + (size_t)b * MODN + l * 1536;
#pragma unroll
      for (int j = 0; j < 6; ++j) modv[j * D_ + tid] = (float)mrow[j * D_ + tid];
    }
    __syncthreads();

    // (2) h = LN(x)*(1+sc1)+sh1  -> hb bf16
    ln_mod(xf, hb, HS, modv + 256, modv + 0, wave, lane);
    __syncthreads();

    // (3) qkv = hb @ Wq^T : three uniform 64-tile passes (q, k, v^T)
    for (int ti = wave; ti < 64; ti += 8) {           // q (scaled by DH^-0.5)
      int mt = ti >> 4, nt = ti & 15;
      v8f acc = gemm_tile(hb, mt * 16, HS, 0, Wq, nt * 16, D_, 0, D_, lane);
      int n = nt * 16 + lane15;
      float bias = qkv_b[l * 3 * D_ + n];
#pragma unroll
      for (int r = 0; r < 8; ++r)
        qb[(mt * 16 + r + 8 * half) * HS + n] = (bf16)((acc[r] + bias) * 0.125f);
    }
    for (int ti = wave; ti < 64; ti += 8) {           // k
      int mt = ti >> 4, nt = ti & 15;
      v8f acc = gemm_tile(hb, mt * 16, HS, 0, Wq, 256 + nt * 16, D_, 0, D_, lane);
      int n = nt * 16 + lane15;
      float bias = qkv_b[l * 3 * D_ + 256 + n];
#pragma unroll
      for (int r = 0; r < 8; ++r)
        kb[(mt * 16 + r + 8 * half) * HS + n] = (bf16)(acc[r] + bias);
    }
    for (int ti = wave; ti < 64; ti += 8) {           // v stored transposed [d][s]
      int mt = ti >> 4, nt = ti & 15;
      v8f acc = gemm_tile(hb, mt * 16, HS, 0, Wq, 512 + nt * 16, D_, 0, D_, lane);
      int n = nt * 16 + lane15;
      float bias = qkv_b[l * 3 * D_ + 512 + n];
#pragma unroll
      for (int r = 0; r < 8; ++r)
        vt[n * VS + (mt * 16 + r + 8 * half)] = (bf16)(acc[r] + bias);
    }
    __syncthreads();

    // (4) scores = q @ k^T per head, softmax in-register -> probs bf16
    for (int u = wave; u < 16; u += 8) {
      int head = u >> 2, rb = u & 3, hd = head * DH_;
      v8f a0 = {}, a1 = {}, a2 = {}, a3 = {};
      for (int kk = 0; kk < DH_; kk += 32) {
        v16bf a = load_a(qb, rb * 16, HS, hd + kk, lane);
        a0 = wmma_bf16(a, load_b(kb, 0,  HS, hd + kk, lane), a0);
        a1 = wmma_bf16(a, load_b(kb, 16, HS, hd + kk, lane), a1);
        a2 = wmma_bf16(a, load_b(kb, 32, HS, hd + kk, lane), a2);
        a3 = wmma_bf16(a, load_b(kb, 48, HS, hd + kk, lane), a3);
      }
      bf16* prow = pr + head * (S_ * PS);
#pragma unroll
      for (int r = 0; r < 8; ++r) {
        float v0 = a0[r], v1 = a1[r], v2 = a2[r], v3 = a3[r];
        float mx = fmaxf(fmaxf(v0, v1), fmaxf(v2, v3));
#pragma unroll
        for (int off = 8; off; off >>= 1) mx = fmaxf(mx, __shfl_xor(mx, off, 32));
        float e0 = __expf(v0 - mx), e1 = __expf(v1 - mx);
        float e2 = __expf(v2 - mx), e3 = __expf(v3 - mx);
        float sm = e0 + e1 + e2 + e3;
#pragma unroll
        for (int off = 8; off; off >>= 1) sm += __shfl_xor(sm, off, 32);
        float inv = 1.f / sm;
        int s = rb * 16 + r + 8 * half;
        prow[s * PS +  0 + lane15] = (bf16)(e0 * inv);
        prow[s * PS + 16 + lane15] = (bf16)(e1 * inv);
        prow[s * PS + 32 + lane15] = (bf16)(e2 * inv);
        prow[s * PS + 48 + lane15] = (bf16)(e3 * inv);
      }
    }
    __syncthreads();

    // (5) attn_out = probs @ v  -> hb
    for (int ti = wave; ti < 64; ti += 8) {
      int head = ti >> 4, rem = ti & 15, mt = rem >> 2, nt = rem & 3;
      const bf16* prow = pr + head * (S_ * PS);
      v8f acc = gemm_tile(prow, mt * 16, PS, 0, vt, head * DH_ + nt * 16, VS, 0, S_, lane);
      int n = head * DH_ + nt * 16 + lane15;
#pragma unroll
      for (int r = 0; r < 8; ++r)
        hb[(mt * 16 + r + 8 * half) * HS + n] = (bf16)acc[r];
    }
    __syncthreads();

    // (6) x += g1 * (attn_out @ Wo^T + b)
    for (int ti = wave; ti < 64; ti += 8) {
      int mt = ti >> 4, nt = ti & 15;
      v8f acc = gemm_tile(hb, mt * 16, HS, 0, Wo, nt * 16, D_, 0, D_, lane);
      int n = nt * 16 + lane15;
      float bias = op_b[l * D_ + n];
      float g = modv[512 + n];
#pragma unroll
      for (int r = 0; r < 8; ++r) {
        int m = mt * 16 + r + 8 * half;
        xf[m * D_ + n] += g * (acc[r] + bias);
      }
    }
    __syncthreads();

    // (7) h = LN(x)*(1+sc2)+sh2
    ln_mod(xf, hb, HS, modv + 1024, modv + 768, wave, lane);
    __syncthreads();

    // (8) ffn hidden = mish(hb @ W1^T + b1)
    for (int ti = wave; ti < 128; ti += 8) {
      int mt = ti >> 5, nt = ti & 31;
      v8f acc = gemm_tile(hb, mt * 16, HS, 0, W1, nt * 16, D_, 0, D_, lane);
      int n = nt * 16 + lane15;
      float bias = f1_b[l * DF_ + n];
#pragma unroll
      for (int r = 0; r < 8; ++r)
        ffb[(mt * 16 + r + 8 * half) * FS + n] = (bf16)mish_f(acc[r] + bias);
    }
    __syncthreads();

    // (9) x += g2 * (ffb @ W2^T + b2)
    for (int ti = wave; ti < 64; ti += 8) {
      int mt = ti >> 4, nt = ti & 15;
      v8f acc = gemm_tile(ffb, mt * 16, FS, 0, W2, nt * 16, DF_, 0, DF_, lane);
      int n = nt * 16 + lane15;
      float bias = f2_b[l * D_ + n];
      float g = modv[1280 + n];
#pragma unroll
      for (int r = 0; r < 8; ++r) {
        int m = mt * 16 + r + 8 * half;
        xf[m * D_ + n] += g * (acc[r] + bias);
      }
    }
    __syncthreads();
  }

  // final adaLN + projection (modulation precomputed at offset 9216)
  {
    const bf16* mrow = modb + (size_t)b * MODN + 9216;
    modv[tid]       = (float)mrow[tid];
    modv[256 + tid] = (float)mrow[256 + tid];
  }
  __syncthreads();
  ln_mod(xf, hb, HS, modv + 256, modv + 0, wave, lane);
  __syncthreads();

  float* ob = out + (size_t)b * S_ * D_;
  for (int ti = wave; ti < 64; ti += 8) {
    int mt = ti >> 4, nt = ti & 15;
    v8f acc = gemm_tile(hb, mt * 16, HS, 0, fp_wt, nt * 16, D_, 0, D_, lane);
    int n = nt * 16 + lane15;
    float bias = fp_b[n];
#pragma unroll
    for (int r = 0; r < 8; ++r)
      ob[(mt * 16 + r + 8 * half) * D_ + n] = acc[r] + bias;
  }
}

extern "C" void kernel_launch(void* const* d_in, const int* in_sizes, int n_in,
                              void* d_out, int out_size, void* d_ws, size_t ws_size,
                              hipStream_t stream) {
  const float* x_t      = (const float*)d_in[0];
  const int*   t        = (const int*)d_in[1];
  const float* backbone = (const float*)d_in[2];
  const float* emb      = (const float*)d_in[3];
  const float* te_w1    = (const float*)d_in[4];
  const float* te_b1    = (const float*)d_in[5];
  const float* te_w2    = (const float*)d_in[6];
  const float* te_b2    = (const float*)d_in[7];
  const float* cp_w1    = (const float*)d_in[8];
  const float* cp_b1    = (const float*)d_in[9];
  const float* cp_w2    = (const float*)d_in[10];
  const float* cp_b2    = (const float*)d_in[11];
  const float* adaln_w  = (const float*)d_in[12];
  const float* adaln_b  = (const float*)d_in[13];
  const float* qkv_w    = (const float*)d_in[14];
  const float* qkv_b    = (const float*)d_in[15];
  const float* op_w     = (const float*)d_in[16];
  const float* op_b     = (const float*)d_in[17];
  const float* ffn_w1   = (const float*)d_in[18];
  const float* ffn_b1   = (const float*)d_in[19];
  const float* ffn_w2   = (const float*)d_in[20];
  const float* ffn_b2   = (const float*)d_in[21];
  const float* fa_w     = (const float*)d_in[22];
  const float* fa_b     = (const float*)d_in[23];
  const float* fp_w     = (const float*)d_in[24];
  const float* fp_b     = (const float*)d_in[25];

  char* ws = (char*)d_ws;
  size_t off = 0;
  bf16* qkv_wt = (bf16*)(ws + off); off += (size_t)L_ * 3 * D_ * D_ * 2;
  bf16* op_wt  = (bf16*)(ws + off); off += (size_t)L_ * D_ * D_ * 2;
  bf16* f1_wt  = (bf16*)(ws + off); off += (size_t)L_ * DF_ * D_ * 2;
  bf16* f2_wt  = (bf16*)(ws + off); off += (size_t)L_ * D_ * DF_ * 2;
  bf16* fp_wt  = (bf16*)(ws + off); off += (size_t)D_ * D_ * 2;
  bf16* te_w1t = (bf16*)(ws + off); off += (size_t)D_ * 4 * D_ * 2;
  bf16* te_w2t = (bf16*)(ws + off); off += (size_t)4 * D_ * D_ * 2;
  bf16* cp_w1t = (bf16*)(ws + off); off += (size_t)DM_ * D_ * 2;
  bf16* cp_w2t = (bf16*)(ws + off); off += (size_t)D_ * D_ * 2;
  bf16* wmod_t = (bf16*)(ws + off); off += (size_t)MODN * D_ * 2;
  float* biasmod = (float*)(ws + off); off += (size_t)MODN * 4;
  bf16* teg  = (bf16*)(ws + off); off += (size_t)B_ * D_ * 2;
  bf16* cm   = (bf16*)(ws + off); off += (size_t)B_ * DM_ * 2;
  bf16* teh  = (bf16*)(ws + off); off += (size_t)B_ * 4 * D_ * 2;
  float* cbuf = (float*)(ws + off); off += (size_t)B_ * D_ * 4;
  bf16* cph  = (bf16*)(ws + off); off += (size_t)B_ * D_ * 2;
  bf16* mcb  = (bf16*)(ws + off); off += (size_t)B_ * D_ * 2;
  bf16* modb = (bf16*)(ws + off); off += (size_t)B_ * MODN * 2;

  // ---- weight conversion: fp32 [K][N] -> bf16 [N][K] ----
  {
    int tq = L_ * D_ * 3 * D_;
    wt_prep<<<(tq + 255) / 256, 256, 0, stream>>>(qkv_w, qkv_wt, D_, 3 * D_, tq);
    int to = L_ * D_ * D_;
    wt_prep<<<(to + 255) / 256, 256, 0, stream>>>(op_w, op_wt, D_, D_, to);
    int t1 = L_ * D_ * DF_;
    wt_prep<<<(t1 + 255) / 256, 256, 0, stream>>>(ffn_w1, f1_wt, D_, DF_, t1);
    int t2 = L_ * DF_ * D_;
    wt_prep<<<(t2 + 255) / 256, 256, 0, stream>>>(ffn_w2, f2_wt, DF_, D_, t2);
    int tf = D_ * D_;
    wt_prep<<<(tf + 255) / 256, 256, 0, stream>>>(fp_w, fp_wt, D_, D_, tf);
    int ta = D_ * 4 * D_;
    wt_prep<<<(ta + 255) / 256, 256, 0, stream>>>(te_w1, te_w1t, D_, 4 * D_, ta);
    wt_prep<<<(ta + 255) / 256, 256, 0, stream>>>(te_w2, te_w2t, 4 * D_, D_, ta);
    int tc = DM_ * D_;
    wt_prep<<<(tc + 255) / 256, 256, 0, stream>>>(cp_w1, cp_w1t, DM_, D_, tc);
    wt_prep<<<(tf + 255) / 256, 256, 0, stream>>>(cp_w2, cp_w2t, D_, D_, tf);
    int tm = MODN * D_;
    modw_prep<<<(tm + 255) / 256, 256, 0, stream>>>(adaln_w, adaln_b, fa_w, fa_b,
                                                    wmod_t, biasmod);
  }

  // ---- conditioning pipeline as batched WMMA GEMMs ----
  gather_kernel<<<B_, 256, 0, stream>>>(t, emb, backbone, teg, cm);
  // teh = mish(teg @ te_w1 + b1)            [B,1024]
  gemm_bf16<<<dim3(8, B_ / 64), 256, 0, stream>>>(teg, te_w1t, te_b1, nullptr,
                                                  teh, nullptr, D_, 4 * D_, 1);
  // cbuf = teh @ te_w2 + b2 (f32)           [B,256]
  gemm_bf16<<<dim3(2, B_ / 64), 256, 0, stream>>>(teh, te_w2t, te_b2, nullptr,
                                                  nullptr, cbuf, 4 * D_, D_, 0);
  // cph = mish(cm @ cp_w1 + b1)             [B,256]
  gemm_bf16<<<dim3(2, B_ / 64), 256, 0, stream>>>(cm, cp_w1t, cp_b1, nullptr,
                                                  cph, nullptr, DM_, D_, 1);
  // mcb = mish(cph @ cp_w2 + b2 + cbuf)     [B,256]
  gemm_bf16<<<dim3(2, B_ / 64), 256, 0, stream>>>(cph, cp_w2t, cp_b2, cbuf,
                                                  mcb, nullptr, D_, D_, 1);
  // modb = mcb @ Wmod + biasmod             [B,9728]
  gemm_bf16<<<dim3(MODN / 128, B_ / 64), 256, 0, stream>>>(mcb, wmod_t, biasmod, nullptr,
                                                           modb, nullptr, D_, MODN, 0);

  hipFuncSetAttribute((const void*)dit_main,
                      hipFuncAttributeMaxDynamicSharedMemorySize, SMEM_BYTES);
  dit_main<<<B_, 256, SMEM_BYTES, stream>>>(x_t, modb,
                                            qkv_wt, qkv_b, op_wt, op_b,
                                            f1_wt, ffn_b1, f2_wt, ffn_b2,
                                            fp_wt, fp_b, (float*)d_out);
}
